// EnhancedMLPDenoisingVAE_83580063580747
// MI455X (gfx1250) — matrix operand order repaired and checked by reading, not compiled, not run on hardware
//
#include <hip/hip_runtime.h>
#include <hip/hip_bf16.h>
#include <stdint.h>

// ---------------------------------------------------------------------------
// EnhancedMLPDenoisingVAE forward for gfx1250 (MI455X), bf16 WMMA pipeline.
// Round 3: async-to-LDS double-buffered GEMM; builtin prototype fixed to
// int4 addrspace(1)* / addrspace(3)* per clang diagnostic.
// ---------------------------------------------------------------------------

typedef __attribute__((ext_vector_type(16))) __bf16 v16bf;
typedef __attribute__((ext_vector_type(8)))  __bf16 v8bf;
typedef __attribute__((ext_vector_type(8)))  float  v8f;
typedef __attribute__((ext_vector_type(4)))  int    v4i;

#define BSZ 32768
#define DIM 768
#define HID 1024
#define LAT 256
#define CTXM 32

#define BM 128
#define BN 128
#define BK 32
#define LDA 40   // padded LDS row stride in bf16 (80B: 16B-aligned, conflict-free)

#if defined(__HIP_DEVICE_COMPILE__) && defined(__has_builtin)
#if __has_builtin(__builtin_amdgcn_global_load_async_to_lds_b128)
#define HAVE_ASYNC_LDS 1
#endif
#endif
#ifndef HAVE_ASYNC_LDS
#define HAVE_ASYNC_LDS 0
#endif

#if HAVE_ASYNC_LDS
typedef __attribute__((address_space(1))) v4i gas_v4i;   // global int4
typedef __attribute__((address_space(3))) v4i las_v4i;   // LDS int4

__device__ __forceinline__ void async_copy16B(const __bf16* g, __bf16* l)
{
    // GLOBAL_LOAD_ASYNC_TO_LDS_B128: per-lane 16B global -> LDS, ASYNCcnt.
    __builtin_amdgcn_global_load_async_to_lds_b128(
        (gas_v4i*)(g), (las_v4i*)(l), 0, 0);
}
__device__ __forceinline__ void async_wait0()
{
#if __has_builtin(__builtin_amdgcn_s_wait_asynccnt)
    __builtin_amdgcn_s_wait_asynccnt(0);
#else
    asm volatile("s_wait_asynccnt 0x0" ::: "memory");
#endif
}
#endif

// ---------------------------------------------------------------------------
// GEMM: C[B,N] = epi( A_bf16[B,K] @ W_bf16[N,K]^T + bias )
// epi == 0 : raw fp32 output ; epi == 1 : leaky-relu(0.2), bf16 output
// ---------------------------------------------------------------------------
__global__ __launch_bounds__(256)
void gemm_bf16_wmma(const __bf16* __restrict__ A,
                    const __bf16* __restrict__ W,
                    const float*  __restrict__ bias,
                    float*        __restrict__ Cf,
                    __bf16*       __restrict__ Cb,
                    int K, int N, int epi)
{
    __shared__ __bf16 As[2][BM * LDA];

    const int tid  = threadIdx.x;
    const int lane = tid & 31;
    const int wave = tid >> 5;
    const int wm   = wave & 1;        // 0..1  -> 64-row slab
    const int wn   = wave >> 1;       // 0..3  -> 32-col slab
    const int l15  = lane & 15;
    const int lh   = lane >> 4;       // 0/1

    const int bm = blockIdx.x * BM;
    const int bn = blockIdx.y * BN;

    // cooperative A-tile mapping: 2 threads per row, 16 bf16 (32B) each
    const int arow = tid >> 1;
    const int acol = (tid & 1) * 16;

    v8f acc[4][2];
#pragma unroll
    for (int s = 0; s < 4; ++s)
#pragma unroll
        for (int n = 0; n < 2; ++n)
#pragma unroll
            for (int j = 0; j < 8; ++j) acc[s][n][j] = 0.0f;

    const __bf16* wrow[2];
#pragma unroll
    for (int n = 0; n < 2; ++n)
        wrow[n] = W + (size_t)(bn + wn * 32 + n * 16 + l15) * (size_t)K + lh * 16;

    const __bf16* ag = A + (size_t)(bm + arow) * (size_t)K + acol;
    const int nt = K / BK;

#if HAVE_ASYNC_LDS
    // ---- async double-buffered pipeline: 1 barrier / K-step ----
    {
        __bf16* l0 = &As[0][arow * LDA + acol];
        async_copy16B(ag,     l0);
        async_copy16B(ag + 8, l0 + 8);
    }
    for (int t = 0; t < nt; ++t) {
        const int cur = t & 1;
        const int k0  = t * BK;
        async_wait0();       // own wave's staged tile complete
        __syncthreads();     // whole block: tile `cur` visible, old reads done
        if (t + 1 < nt) {
            __bf16* ln = &As[cur ^ 1][arow * LDA + acol];
            const __bf16* gn = ag + (size_t)(k0 + BK);
            async_copy16B(gn,     ln);
            async_copy16B(gn + 8, ln + 8);
        }
#else
    for (int t = 0; t < nt; ++t) {
        const int cur = 0;
        const int k0  = t * BK;
        __syncthreads();
        {
            const uint4* g = (const uint4*)(ag + k0);
            uint4* l = (uint4*)&As[0][arow * LDA + acol];
            l[0] = g[0];
            l[1] = g[1];
            if (k0 + BK < K)
                __builtin_prefetch(ag + k0 + BK, 0, 0);   // global_prefetch_b8
        }
        __syncthreads();
#endif
        // B fragments straight from global (weights are L2-resident)
        v16bf bf[2];
#pragma unroll
        for (int n = 0; n < 2; ++n) {
            const __bf16* wp = wrow[n] + k0;
            v8bf b0 = *(const v8bf*)(wp);
            v8bf b1 = *(const v8bf*)(wp + 8);
            bf[n] = __builtin_shufflevector(b0, b1,
                        0,1,2,3,4,5,6,7,8,9,10,11,12,13,14,15);
        }
#pragma unroll
        for (int s = 0; s < 4; ++s) {
            const __bf16* ap = &As[cur][(wm * 64 + s * 16 + l15) * LDA + lh * 16];
            v8bf a0 = *(const v8bf*)(ap);
            v8bf a1 = *(const v8bf*)(ap + 8);
            v16bf af = __builtin_shufflevector(a0, a1,
                        0,1,2,3,4,5,6,7,8,9,10,11,12,13,14,15);
#pragma unroll
            for (int n = 0; n < 2; ++n) {
                acc[s][n] = __builtin_amdgcn_wmma_f32_16x16x32_bf16(
                    false, af, false, bf[n], (short)0, acc[s][n], false, false);
            }
        }
    }

    // epilogue: bias (+ optional lrelu), scatter stores
    float bv[2];
#pragma unroll
    for (int n = 0; n < 2; ++n)
        bv[n] = bias[bn + wn * 32 + n * 16 + l15];

#pragma unroll
    for (int s = 0; s < 4; ++s) {
#pragma unroll
        for (int n = 0; n < 2; ++n) {
            const int col = bn + wn * 32 + n * 16 + l15;
#pragma unroll
            for (int j = 0; j < 8; ++j) {
                const int row = bm + wm * 64 + s * 16 + lh * 8 + j;
                float v = acc[s][n][j] + bv[n];
                if (epi == 1) {
                    v = (v >= 0.0f) ? v : 0.2f * v;
                    Cb[(size_t)row * (size_t)N + col] = (__bf16)v;
                } else {
                    Cf[(size_t)row * (size_t)N + col] = v;
                }
            }
        }
    }
}

// ---------------------------------------------------------------------------
// Row LayerNorm + leaky-relu, fp32 in -> bf16 out. One block per row.
// ---------------------------------------------------------------------------
__global__ __launch_bounds__(256)
void ln_lrelu_bf16(const float* __restrict__ X,
                   const float* __restrict__ g,
                   const float* __restrict__ b,
                   __bf16* __restrict__ Y, int N)
{
    const int row = blockIdx.x;
    const int tid = threadIdx.x;
    const float* xr = X + (size_t)row * (size_t)N;

    float s = 0.0f, ss = 0.0f;
    for (int i = tid; i < N; i += 256) {
        float v = xr[i];
        s += v;
        ss += v * v;
    }
#pragma unroll
    for (int o = 16; o; o >>= 1) {
        s  += __shfl_xor(s,  o);
        ss += __shfl_xor(ss, o);
    }
    __shared__ float rs[8], rss[8], mv[2];
    if ((tid & 31) == 0) { rs[tid >> 5] = s; rss[tid >> 5] = ss; }
    __syncthreads();
    if (tid == 0) {
        float S = 0.0f, SS = 0.0f;
#pragma unroll
        for (int w = 0; w < 8; ++w) { S += rs[w]; SS += rss[w]; }
        float mu  = S / (float)N;
        float var = SS / (float)N - mu * mu;
        mv[0] = mu;
        mv[1] = rsqrtf(var + 1e-5f);
    }
    __syncthreads();
    const float mu = mv[0], rstd = mv[1];
    __bf16* yr = Y + (size_t)row * (size_t)N;
    for (int i = tid; i < N; i += 256) {
        float v = (xr[i] - mu) * rstd * g[i] + b[i];
        v = (v >= 0.0f) ? v : 0.2f * v;
        yr[i] = (__bf16)v;
    }
}

// ---------------------------------------------------------------------------
// z = mu + eps * exp(0.5*logvar)
// ---------------------------------------------------------------------------
__global__ __launch_bounds__(256)
void reparam_kernel(const float* __restrict__ mu,
                    const float* __restrict__ lv,
                    const float* __restrict__ eps,
                    float* __restrict__ z, int n)
{
    int i = blockIdx.x * 256 + threadIdx.x;
    if (i < n) z[i] = mu[i] + eps[i] * expf(0.5f * lv[i]);
}

// ---------------------------------------------------------------------------
// context attention: attn = softmax(z @ ctx^T); z_enh = z + 0.1*attn@ctx
// ---------------------------------------------------------------------------
__global__ __launch_bounds__(256)
void ctx_attn_kernel(const float* __restrict__ Z,
                     const float* __restrict__ ctx,  // [32,256] fp32
                     __bf16* __restrict__ out)
{
    __shared__ float zsh[LAT];
    __shared__ float ash[CTXM];
    const int row = blockIdx.x;
    const int tid = threadIdx.x;
    zsh[tid] = Z[(size_t)row * LAT + tid];
    __syncthreads();
    if (tid < CTXM) {
        float sc = 0.0f;
        for (int k = 0; k < LAT; ++k) sc += zsh[k] * ctx[tid * LAT + k];
        float mx = sc;
#pragma unroll
        for (int o = 16; o; o >>= 1) mx = fmaxf(mx, __shfl_xor(mx, o));
        float e = expf(sc - mx);
        float sum = e;
#pragma unroll
        for (int o = 16; o; o >>= 1) sum += __shfl_xor(sum, o);
        ash[tid] = e / sum;
    }
    __syncthreads();
    float a = 0.0f;
#pragma unroll
    for (int m = 0; m < CTXM; ++m) a += ash[m] * ctx[m * LAT + tid];
    out[(size_t)row * LAT + tid] = (__bf16)(zsh[tid] + 0.1f * a);
}

// ---------------------------------------------------------------------------
// fp32 -> bf16 convert
// ---------------------------------------------------------------------------
__global__ __launch_bounds__(256)
void cvt_f32_bf16(const float* __restrict__ s, __bf16* __restrict__ d, int n)
{
    int i = blockIdx.x * 256 + threadIdx.x;
    if (i < n) d[i] = (__bf16)s[i];
}

// ---------------------------------------------------------------------------
// host side
// ---------------------------------------------------------------------------
extern "C" void kernel_launch(void* const* d_in, const int* in_sizes, int n_in,
                              void* d_out, int out_size, void* d_ws, size_t ws_size,
                              hipStream_t stream)
{
    (void)in_sizes; (void)n_in; (void)out_size; (void)ws_size;

    const float* x      = (const float*)d_in[0];
    const float* eps    = (const float*)d_in[1];
    const float* enc_w1 = (const float*)d_in[2];
    const float* enc_b1 = (const float*)d_in[3];
    const float* ln1_g  = (const float*)d_in[4];
    const float* ln1_b  = (const float*)d_in[5];
    const float* enc_w2 = (const float*)d_in[6];
    const float* enc_b2 = (const float*)d_in[7];
    const float* ln2_g  = (const float*)d_in[8];
    const float* ln2_b  = (const float*)d_in[9];
    const float* mu_w   = (const float*)d_in[10];
    const float* mu_b   = (const float*)d_in[11];
    const float* lv_w   = (const float*)d_in[12];
    const float* lv_b   = (const float*)d_in[13];
    const float* di_w   = (const float*)d_in[14];
    const float* di_b   = (const float*)d_in[15];
    const float* dec_w1 = (const float*)d_in[16];
    const float* dec_b1 = (const float*)d_in[17];
    const float* dln1_g = (const float*)d_in[18];
    const float* dln1_b = (const float*)d_in[19];
    const float* dec_w2 = (const float*)d_in[20];
    const float* dec_b2 = (const float*)d_in[21];
    const float* dln2_g = (const float*)d_in[22];
    const float* dln2_b = (const float*)d_in[23];
    const float* dec_w3 = (const float*)d_in[24];
    const float* dec_b3 = (const float*)d_in[25];
    const float* ctx    = (const float*)d_in[26];

    float* out   = (float*)d_out;
    float* mu_o  = out + (size_t)BSZ * DIM;
    float* lv_o  = mu_o + (size_t)BSZ * LAT;

    // ---- workspace carve-up ----
    char* ws = (char*)d_ws;
    size_t off = 0;
    auto carve = [&](size_t bytes) { char* p = ws + off; off += (bytes + 255) & ~(size_t)255; return p; };

    __bf16* w1b  = (__bf16*)carve((size_t)HID * DIM * 2);
    __bf16* w2b  = (__bf16*)carve((size_t)HID * HID * 2);
    __bf16* muwb = (__bf16*)carve((size_t)LAT * HID * 2);
    __bf16* lvwb = (__bf16*)carve((size_t)LAT * HID * 2);
    __bf16* diwb = (__bf16*)carve((size_t)HID * LAT * 2);
    __bf16* dw1b = (__bf16*)carve((size_t)HID * HID * 2);
    __bf16* dw2b = (__bf16*)carve((size_t)2 * HID * HID * 2);
    __bf16* dw3b = (__bf16*)carve((size_t)DIM * 2 * HID * 2);

    float*  F32  = (float*) carve((size_t)BSZ * 2 * HID * 4);   // raw pre-LN
    __bf16* A16a = (__bf16*)carve((size_t)BSZ * 2 * HID * 2);   // activations ping
    __bf16* A16b = (__bf16*)carve((size_t)BSZ * 2 * HID * 2);   // activations pong
    float*  Z32  = (float*) carve((size_t)BSZ * LAT * 4);       // z

    auto cvt = [&](const float* s, __bf16* d, size_t n) {
        cvt_f32_bf16<<<(unsigned)((n + 255) / 256), 256, 0, stream>>>(s, d, (int)n);
    };
    auto gemm = [&](const __bf16* Aact, const __bf16* Wb, const float* bias,
                    void* Cout, int K, int N, int epi) {
        dim3 grid(BSZ / BM, (unsigned)(N / BN));
        gemm_bf16_wmma<<<grid, 256, 0, stream>>>(Aact, Wb, bias,
                                                 (float*)Cout, (__bf16*)Cout, K, N, epi);
    };
    auto ln = [&](const float* X, const float* g, const float* b, __bf16* Y, int N) {
        ln_lrelu_bf16<<<BSZ, 256, 0, stream>>>(X, g, b, Y, N);
    };

    // ---- weight conversion (cheap, every call; deterministic) ----
    cvt(enc_w1, w1b,  (size_t)HID * DIM);
    cvt(enc_w2, w2b,  (size_t)HID * HID);
    cvt(mu_w,   muwb, (size_t)LAT * HID);
    cvt(lv_w,   lvwb, (size_t)LAT * HID);
    cvt(di_w,   diwb, (size_t)HID * LAT);
    cvt(dec_w1, dw1b, (size_t)HID * HID);
    cvt(dec_w2, dw2b, (size_t)2 * HID * HID);
    cvt(dec_w3, dw3b, (size_t)DIM * 2 * HID);

    // ---- encoder ----
    cvt(x, A16a, (size_t)BSZ * DIM);                       // x -> bf16
    gemm(A16a, w1b, enc_b1, F32, DIM, HID, 0);             // h1 raw
    ln(F32, ln1_g, ln1_b, A16b, HID);                      // h1 bf16
    gemm(A16b, w2b, enc_b2, F32, HID, HID, 0);             // h2 raw
    ln(F32, ln2_g, ln2_b, A16a, HID);                      // h2 bf16
    gemm(A16a, muwb, mu_b, mu_o, HID, LAT, 0);             // mu  -> d_out
    gemm(A16a, lvwb, lv_b, lv_o, HID, LAT, 0);             // lv  -> d_out

    // ---- reparameterize + context attention ----
    reparam_kernel<<<(BSZ * LAT) / 256, 256, 0, stream>>>(mu_o, lv_o, eps, Z32, BSZ * LAT);
    ctx_attn_kernel<<<BSZ, 256, 0, stream>>>(Z32, ctx, A16b);  // z_enh bf16

    // ---- decoder ----
    gemm(A16b, diwb, di_b, A16a, LAT, HID, 1);             // d1 = lrelu(...), bf16
    gemm(A16a, dw1b, dec_b1, F32, HID, HID, 0);            // d2 raw
    ln(F32, dln1_g, dln1_b, A16b, HID);                    // d2 bf16
    gemm(A16b, dw2b, dec_b2, F32, HID, 2 * HID, 0);        // d3 raw [B,2048]
    ln(F32, dln2_g, dln2_b, A16a, 2 * HID);                // d3 bf16
    gemm(A16a, dw3b, dec_b3, out, 2 * HID, DIM, 0);        // reconstruction -> d_out
}
